// SimpleNetwork_22746146800187
// MI455X (gfx1250) — compile-verified, matrix operand order checked by reading, not run on hardware
//
#include <hip/hip_runtime.h>
#include <math.h>

// ---------------------------------------------------------------------------
// Equivariant GNN (3 layers) for MI455X / gfx1250 (wave32, WMMA).
// Edge radial-MLP runs on v_wmma_f32_16x16x32_f16 (16-edge M tiles).
// Weights pre-swizzled into B-operand lane order -> contiguous b128 loads;
// staged once per block in LDS; bias folded in as a constant-1 K channel.
// 4 tiles per wave amortize the weight staging; per-wave LDS regions need
// only wave-level scheduling barriers (LDS is in-order within a wave).
// Message scatter uses hardware f32 global atomics (L2-resident agg buffers).
// ---------------------------------------------------------------------------

typedef __attribute__((ext_vector_type(16))) _Float16 v16h;
typedef __attribute__((ext_vector_type(8)))  float    v8f;

#define MUL 16
#define NB  10
#define WPB 4   // waves per block in edge kernel (128 threads)
#define TPW 4   // 16-edge tiles per wave

// K index held by (lane, vgpr j, half b) for a 16-bit 16x32 A/B operand
// (ISA 7.12.2): lanes 0-15: K = {0..7, 16..23}; lanes 16-31: +8.
__device__ __forceinline__ int kPat(int lane, int j, int b) {
  int base = (j < 4) ? (2 * j) : (16 + 2 * (j - 4));
  return base + b + ((lane & 16) ? 8 : 0);
}

__device__ __forceinline__ float siluf(float v)    { return v / (1.f + __expf(-v)); }
__device__ __forceinline__ float sigmoidf_(float v){ return 1.f / (1.f + __expf(-v)); }

// ---------------------------------------------------------------------------
// Pad + convert + swizzle MLP weights to f16 B-operand layout, once.
//  w1s[l][nt=7][lane=32][h=16]   (K=32 x N=112 tiles; K=31 row carries b1)
//  w2s[l][nt=3][kc=4][lane=32][h=16]  (K=128 x N=48 tiles)
// ---------------------------------------------------------------------------
__global__ void prep_weights_kernel(const float* __restrict__ w1, const float* __restrict__ b1,
                                    const float* __restrict__ w2,
                                    _Float16* __restrict__ w1s, _Float16* __restrict__ w2s) {
  int i = blockIdx.x * blockDim.x + threadIdx.x;
  if (i < 3 * 7 * 512) {
    int l = i / 3584, r = i % 3584;
    int nt = r / 512, q = r % 512, lane = q / 16, h = q % 16;
    int K = kPat(lane, h >> 1, h & 1);
    int n = nt * 16 + (lane & 15);
    float v = 0.f;
    if (n < 100) {
      if (K < NB)        v = w1[l * 1000 + K * 100 + n];
      else if (K == 31)  v = b1[l * 100 + n];   // bias via constant-1 emb channel
    }
    w1s[i] = (_Float16)v;
  }
  if (i < 3 * 12 * 512) {
    int l = i / 6144, r = i % 6144;
    int t = r / 512, nt = t / 4, kc = t % 4, q = r % 512, lane = q / 16, h = q % 16;
    int K = kc * 32 + kPat(lane, h >> 1, h & 1);
    int n = nt * 16 + (lane & 15);
    w2s[i] = (K < 100) ? (_Float16)w2[l * 4800 + K * 48 + n] : (_Float16)0.f;
  }
}

// ---------------------------------------------------------------------------
// p0 = x0 @ P0[l]   (N x 16 @ 16 x 16)
// ---------------------------------------------------------------------------
__global__ __launch_bounds__(256) void p0_kernel(const float* __restrict__ x0,
                                                 const float* __restrict__ P0,
                                                 float* __restrict__ p0, int N) {
  __shared__ float Ps[256];
  if (threadIdx.x < 256) Ps[threadIdx.x] = P0[threadIdx.x];
  __syncthreads();
  int idx = blockIdx.x * blockDim.x + threadIdx.x;
  if (idx >= N * MUL) return;
  int n = idx >> 4, m = idx & 15;
  const float* xr = x0 + n * MUL;
  float acc = 0.f;
#pragma unroll
  for (int c = 0; c < MUL; ++c) acc = fmaf(xr[c], Ps[c * MUL + m], acc);
  p0[idx] = acc;
}

// ---------------------------------------------------------------------------
// Edge kernel: TPW 16-edge tiles per wave, 4 waves / 128-thread block.
// ---------------------------------------------------------------------------
__global__ __launch_bounds__(128) void edge_kernel(
    const float* __restrict__ pos, const int* __restrict__ esrc, const int* __restrict__ edst,
    const float* __restrict__ p0, float* __restrict__ agg,
    const _Float16* __restrict__ w1s, const _Float16* __restrict__ w2s,
    int E, int numTiles) {
  __shared__ float    sh_s  [WPB][16][8];
  __shared__ float    p0e_s [WPB][16][16];
  __shared__ int      dst_s [WPB][16];
  __shared__ int      src_s [WPB][16];
  __shared__ _Float16 embA_s[WPB][32][16];     // A operand layout, GEMM1
  __shared__ _Float16 hidA_s[WPB][4][32][16];  // A operand layout, GEMM2 (4 K-chunks)
  __shared__ _Float16 w1s_s[7 * 512];          // 7 B tiles, GEMM1
  __shared__ _Float16 w2s_s[12 * 512];         // 12 B tiles, GEMM2

  const int tid  = threadIdx.x;
  const int wv   = tid >> 5;
  const int lane = tid & 31;
  const int n    = lane & 15;
  const int hi   = lane >> 4;

  // ---- cooperative weight staging (whole block, once) ----
  {
    const int4* g1 = (const int4*)w1s;
    int4*       l1 = (int4*)w1s_s;
#pragma unroll 2
    for (int i = tid; i < (7 * 512) / 8; i += 128) l1[i] = g1[i];
    const int4* g2 = (const int4*)w2s;
    int4*       l2 = (int4*)w2s_s;
#pragma unroll 2
    for (int i = tid; i < (12 * 512) / 8; i += 128) l2[i] = g2[i];
  }
  __syncthreads();

  const int tile0 = (blockIdx.x * WPB + wv) * TPW;

#pragma unroll 1
  for (int t = 0; t < TPW; ++t) {
    int tile = tile0 + t;
    const bool valid = tile < numTiles;
    if (!valid) tile = numTiles - 1;
    const int base = tile * 16;

    // ---- phase 1: edge geometry + radial embed written in A-operand order ----
    if (lane < 16) {
      int e = base + lane; if (e >= E) e = E - 1;
      int s = esrc[e], d = edst[e];
      src_s[wv][lane] = s; dst_s[wv][lane] = d;
      __builtin_prefetch(p0 + (size_t)s * MUL, 0, 3);
      float vx = pos[3 * s + 0] - pos[3 * d + 0];
      float vy = pos[3 * s + 1] - pos[3 * d + 1];
      float vz = pos[3 * s + 2] - pos[3 * d + 2];
      float r  = sqrtf(vx * vx + vy * vy + vz * vz);
      float iv = 1.f / (r + 1e-12f);
      float ux = vx * iv, uy = vy * iv, uz = vz * iv;
      const float SQ3 = 1.73205081f, SQ15 = 3.87298335f, SQ5H = 1.11803399f;
      sh_s[wv][lane][0] = SQ3 * ux;
      sh_s[wv][lane][1] = SQ3 * uy;
      sh_s[wv][lane][2] = SQ3 * uz;
      sh_s[wv][lane][3] = SQ15 * ux * uy;
      sh_s[wv][lane][4] = SQ15 * uy * uz;
      sh_s[wv][lane][5] = SQ5H * (3.f * uz * uz - 1.f);
      sh_s[wv][lane][6] = SQ15 * ux * uz;
      sh_s[wv][lane][7] = 0.5f * SQ15 * (ux * ux - uy * uy);
      const float SQNB = 3.16227766f;  // sqrt(10)
#pragma unroll
      for (int k = 0; k < 32; ++k) {
        float val;
        if (k < NB) {
          float dd = (r - (k + 0.5f) * 0.2f) * 5.0f;  // (r - center)/step, step=0.2
          val = (fabsf(dd) < 1.f) ? __cosf(1.57079633f * dd) * SQNB : 0.f;
        } else {
          val = (k == 31) ? 1.f : 0.f;  // constant-1 channel carries the bias
        }
        // A layout scatter: lane = M + 16*((k>>3)&1), half = (k&7) + 8*((k>>4)&1)
        embA_s[wv][lane + 16 * ((k >> 3) & 1)][(k & 7) + 8 * ((k >> 4) & 1)] = (_Float16)val;
      }
    }
    __builtin_amdgcn_wave_barrier();  // per-wave LDS: DS is in-order within a wave

    // ---- phase 2: vectorized p0[src] gather + zero hid pad cols 112..127 ----
    {
      int i = lane & 15, half = lane >> 4;
      int s = src_s[wv][i];
      const float4* pr = (const float4*)(p0 + (size_t)s * MUL + half * 8);
      float4 a = pr[0], b = pr[1];
      float4* dst4 = (float4*)&p0e_s[wv][i][half * 8];
      dst4[0] = a; dst4[1] = b;
      int4 z = {0, 0, 0, 0};
      *(int4*)&hidA_s[wv][3][lane][8] = z;  // cols 112..127 of the K=96..127 chunk
    }
    __builtin_amdgcn_wave_barrier();

    // ---- GEMM1: hidden(16x112) = [emb|1](16x32) @ [W1;b1](32x112), then silu ----
    v16h a1 = *(const v16h*)&embA_s[wv][lane][0];
    v8f c1[7];
#pragma unroll
    for (int nt = 0; nt < 7; ++nt) {
#pragma unroll
      for (int j = 0; j < 8; ++j) c1[nt][j] = 0.f;
    }
#pragma unroll
    for (int nt = 0; nt < 7; ++nt) {
      v16h b = *(const v16h*)&w1s_s[nt * 512 + lane * 16];
      c1[nt] = __builtin_amdgcn_wmma_f32_16x16x32_f16(false, a1, false, b,
                                                      (short)0, c1[nt], false, false);
    }
    // silu + scatter into GEMM2 A-operand layout. C layout: row M=j+8*hi, col nt*16+n.
#pragma unroll
    for (int nt = 0; nt < 7; ++nt) {
      int col = nt * 16 + n;
      int kc = col >> 5, kl = col & 31;
      int lsel = ((kl >> 3) & 1) << 4;
      int h = (kl & 7) + 8 * ((kl >> 4) & 1);
#pragma unroll
      for (int j = 0; j < 8; ++j)
        hidA_s[wv][kc][(j + 8 * hi) + lsel][h] = (_Float16)siluf(c1[nt][j]);
    }
    __builtin_amdgcn_wave_barrier();

    // ---- GEMM2: w(16x48) = hidden(16x128) @ W2(128x48) ----
    v8f cw[3];
#pragma unroll
    for (int nt = 0; nt < 3; ++nt)
#pragma unroll
      for (int j = 0; j < 8; ++j) cw[nt][j] = 0.f;
#pragma unroll
    for (int kc = 0; kc < 4; ++kc) {
      v16h a2 = *(const v16h*)&hidA_s[wv][kc][lane][0];
#pragma unroll
      for (int nt = 0; nt < 3; ++nt) {
        v16h b2 = *(const v16h*)&w2s_s[(nt * 4 + kc) * 512 + lane * 16];
        cw[nt] = __builtin_amdgcn_wmma_f32_16x16x32_f16(false, a2, false, b2,
                                                        (short)0, cw[nt], false, false);
      }
    }

    // ---- message scatter: lane owns channel c = n, rows j+8*hi ----
    if (valid) {
      const float invsq = 0.176776695f;  // 1/sqrt(32)
#pragma unroll
      for (int j = 0; j < 8; ++j) {
        int row = j + 8 * hi;
        if (base + row >= E) continue;
        float pe = p0e_s[wv][row][n];
        float* ag = agg + (size_t)dst_s[wv][row] * 144;
        unsafeAtomicAdd(&ag[n], cw[0][j] * pe * invsq);
        float m1 = cw[1][j] * pe * invsq;
#pragma unroll
        for (int k = 0; k < 3; ++k)
          unsafeAtomicAdd(&ag[16 + n * 3 + k], m1 * sh_s[wv][row][k]);
        float m2 = cw[2][j] * pe * invsq;
#pragma unroll
        for (int k = 0; k < 5; ++k)
          unsafeAtomicAdd(&ag[64 + n * 5 + k], m2 * sh_s[wv][row][3 + k]);
      }
    }
    __builtin_amdgcn_wave_barrier();  // finish reads before next tile overwrites LDS
  }
}

// ---------------------------------------------------------------------------
// Node update: s = x0@Wsc + agg0@Wa ; v1 = x1 mix + agg1 ; v2 = x2 mix + agg2
// x0' = silu(s[:16]); x1' = v1*sig(s[16:32]); x2' = v2*sig(s[32:48])
// ---------------------------------------------------------------------------
__global__ __launch_bounds__(256) void node_update_kernel(
    const float* __restrict__ x0o, const float* __restrict__ x1o, const float* __restrict__ x2o,
    const float* __restrict__ agg,
    const float* __restrict__ Wsc, const float* __restrict__ Wa,
    const float* __restrict__ W1m, const float* __restrict__ W2m,
    float* __restrict__ x0n, float* __restrict__ x1n, float* __restrict__ x2n, int N) {
  __shared__ float sWsc[768], sWa[768], sW1[256], sW2[256];
  for (int i = threadIdx.x; i < 768; i += 256) { sWsc[i] = Wsc[i]; sWa[i] = Wa[i]; }
  if (threadIdx.x < 256) { sW1[threadIdx.x] = W1m[threadIdx.x]; sW2[threadIdx.x] = W2m[threadIdx.x]; }
  __syncthreads();
  int idx = blockIdx.x * blockDim.x + threadIdx.x;
  if (idx >= N * MUL) return;
  int nd = idx >> 4, m = idx & 15;
  const float* x0r = x0o + nd * MUL;
  const float* ag  = agg + (size_t)nd * 144;
  float s0 = 0.f, s1 = 0.f, s2 = 0.f;
#pragma unroll
  for (int c = 0; c < MUL; ++c) {
    float xc = x0r[c], ac = ag[c];
    s0 = fmaf(xc, sWsc[c * 48 + m],      fmaf(ac, sWa[c * 48 + m],      s0));
    s1 = fmaf(xc, sWsc[c * 48 + 16 + m], fmaf(ac, sWa[c * 48 + 16 + m], s1));
    s2 = fmaf(xc, sWsc[c * 48 + 32 + m], fmaf(ac, sWa[c * 48 + 32 + m], s2));
  }
  float v1[3], v2[5];
#pragma unroll
  for (int k = 0; k < 3; ++k) v1[k] = ag[16 + m * 3 + k];
#pragma unroll
  for (int k = 0; k < 5; ++k) v2[k] = ag[64 + m * 5 + k];
  const float* x1r = x1o + (size_t)nd * 48;
  const float* x2r = x2o + (size_t)nd * 80;
#pragma unroll
  for (int c = 0; c < MUL; ++c) {
    float w1c = sW1[c * 16 + m], w2c = sW2[c * 16 + m];
#pragma unroll
    for (int k = 0; k < 3; ++k) v1[k] = fmaf(x1r[c * 3 + k], w1c, v1[k]);
#pragma unroll
    for (int k = 0; k < 5; ++k) v2[k] = fmaf(x2r[c * 5 + k], w2c, v2[k]);
  }
  float g1 = sigmoidf_(s1), g2 = sigmoidf_(s2);
  x0n[nd * MUL + m] = siluf(s0);
#pragma unroll
  for (int k = 0; k < 3; ++k) x1n[(size_t)nd * 48 + m * 3 + k] = v1[k] * g1;
#pragma unroll
  for (int k = 0; k < 5; ++k) x2n[(size_t)nd * 80 + m * 5 + k] = v2[k] * g2;
}

// ---------------------------------------------------------------------------
// pooled = sum_n (x0[n] . W_out) / sqrt(N)
// ---------------------------------------------------------------------------
__global__ __launch_bounds__(256) void out_kernel(const float* __restrict__ x0,
                                                  const float* __restrict__ Wout,
                                                  float* __restrict__ out, int N) {
  __shared__ float red[256];
  int nd = blockIdx.x * blockDim.x + threadIdx.x;
  float acc = 0.f;
  if (nd < N) {
    const float* xr = x0 + (size_t)nd * MUL;
#pragma unroll
    for (int c = 0; c < MUL; ++c) acc = fmaf(xr[c], Wout[c], acc);
  }
  red[threadIdx.x] = acc;
  __syncthreads();
  for (int s = 128; s > 0; s >>= 1) {
    if (threadIdx.x < s) red[threadIdx.x] += red[threadIdx.x + s];
    __syncthreads();
  }
  if (threadIdx.x == 0) unsafeAtomicAdd(out, red[0] * rsqrtf((float)N));
}

// ---------------------------------------------------------------------------
extern "C" void kernel_launch(void* const* d_in, const int* in_sizes, int n_in,
                              void* d_out, int out_size, void* d_ws, size_t ws_size,
                              hipStream_t stream) {
  const float* pos    = (const float*)d_in[0];
  const float* x      = (const float*)d_in[1];
  const int*   esrc   = (const int*)d_in[2];
  const int*   edst   = (const int*)d_in[3];
  const float* mlp_w1 = (const float*)d_in[5];
  const float* mlp_b1 = (const float*)d_in[6];
  const float* mlp_w2 = (const float*)d_in[7];
  const float* P0     = (const float*)d_in[8];
  const float* Wsc    = (const float*)d_in[9];
  const float* Wa     = (const float*)d_in[10];
  const float* W1m    = (const float*)d_in[11];
  const float* W2m    = (const float*)d_in[12];
  const float* Wout   = (const float*)d_in[13];

  const int N = in_sizes[0] / 3;
  const int E = in_sizes[2];

  // workspace layout (floats)
  float* ws = (float*)d_ws;
  size_t off = 0;
  auto alloc = [&](size_t cnt) { float* p = ws + off; off += cnt; return p; };
  float* x0A = alloc((size_t)N * 16);
  float* x1A = alloc((size_t)N * 48);
  float* x2A = alloc((size_t)N * 80);
  float* x0B = alloc((size_t)N * 16);
  float* x1B = alloc((size_t)N * 48);
  float* x2B = alloc((size_t)N * 80);
  float* p0  = alloc((size_t)N * 16);
  float* agg = alloc((size_t)N * 144);
  _Float16* w1s = (_Float16*)(ws + off); off += (3 * 7 * 512) / 2;    // 10752 halves
  _Float16* w2s = (_Float16*)(ws + off); off += (3 * 12 * 512) / 2;   // 18432 halves
  if (off * sizeof(float) > ws_size) return;  // insufficient scratch

  hipMemsetAsync(d_out, 0, sizeof(float), stream);
  hipMemcpyAsync(x0A, x, (size_t)N * 16 * sizeof(float), hipMemcpyDeviceToDevice, stream);
  hipMemsetAsync(x1A, 0, (size_t)N * 48 * sizeof(float), stream);
  hipMemsetAsync(x2A, 0, (size_t)N * 80 * sizeof(float), stream);

  prep_weights_kernel<<<(3 * 12 * 512 + 255) / 256, 256, 0, stream>>>(
      mlp_w1, mlp_b1, mlp_w2, w1s, w2s);

  float *x0c = x0A, *x1c = x1A, *x2c = x2A;
  float *x0n = x0B, *x1n = x1B, *x2n = x2B;
  const int numTiles = (E + 15) / 16;
  const int wavesNeeded = (numTiles + TPW - 1) / TPW;
  const int edgeBlks = (wavesNeeded + WPB - 1) / WPB;
  const int nodeBlks = (N * 16 + 255) / 256;

  for (int l = 0; l < 3; ++l) {
    hipMemsetAsync(agg, 0, (size_t)N * 144 * sizeof(float), stream);
    p0_kernel<<<nodeBlks, 256, 0, stream>>>(x0c, P0 + l * 256, p0, N);
    edge_kernel<<<edgeBlks, 128, 0, stream>>>(
        pos, esrc, edst, p0, agg,
        w1s + (size_t)l * 7 * 512, w2s + (size_t)l * 12 * 512,
        E, numTiles);
    node_update_kernel<<<nodeBlks, 256, 0, stream>>>(
        x0c, x1c, x2c, agg,
        Wsc + l * 768, Wa + l * 768, W1m + l * 256, W2m + l * 256,
        x0n, x1n, x2n, N);
    float* t;
    t = x0c; x0c = x0n; x0n = t;
    t = x1c; x1c = x1n; x1n = t;
    t = x2c; x2c = x2n; x2n = t;
  }

  out_kernel<<<(N + 255) / 256, 256, 0, stream>>>(x0c, Wout, (float*)d_out, N);
}
